// GATNet_29686813950597
// MI455X (gfx1250) — matrix operand order repaired and checked by reading, not compiled, not run on hardware
//
#include <hip/hip_runtime.h>
#include <cstdint>
#include <cstddef>

// Problem constants (match reference)
#define N_NODES 10000
#define IN_CH   512
#define HID     128
#define HEADS   4
#define OUT_CH  256
#define N_EDGES 160000
#define E_TOT   (N_EDGES + N_NODES)   // edges + self loops
#define MTILES  (N_NODES / 16)        // 625, N_NODES is a multiple of 16

typedef __attribute__((ext_vector_type(16))) __bf16 v16bf;
typedef __attribute__((ext_vector_type(8)))  float  v8f;

static __device__ __forceinline__ unsigned short f2bf(float f) {
  // round-to-nearest-even fp32 -> bf16
  unsigned int u = __float_as_uint(f);
  u += 0x7FFFu + ((u >> 16) & 1u);
  return (unsigned short)(u >> 16);
}

// ---------------------------------------------------------------------------
// Conversion kernels
// ---------------------------------------------------------------------------
__global__ void k_cvt_bf16(const float* __restrict__ in,
                           unsigned short* __restrict__ out, int n) {
  int i = blockIdx.x * blockDim.x + threadIdx.x;
  if (i < n) out[i] = f2bf(in[i]);
}

__global__ void k_elu_cvt_bf16(const float* __restrict__ in,
                               unsigned short* __restrict__ out, int n) {
  int i = blockIdx.x * blockDim.x + threadIdx.x;
  if (i < n) {
    float v = in[i];
    v = v > 0.f ? v : (__expf(v) - 1.f);   // ELU, alpha=1
    out[i] = f2bf(v);
  }
}

// W[K,Ncol] fp32 (row-major) -> Wt[Ncol,K] bf16 (row-major, i.e. transposed)
__global__ void k_cvt_w_t(const float* __restrict__ W,
                          unsigned short* __restrict__ Wt, int K, int Ncol) {
  int i = blockIdx.x * blockDim.x + threadIdx.x;
  if (i >= K * Ncol) return;
  int k = i / Ncol, n = i - k * Ncol;
  Wt[(size_t)n * K + k] = f2bf(W[i]);
}

// ---------------------------------------------------------------------------
// bf16 WMMA GEMM: C[M,Ncol] (+= / =) A[M,K] @ Bt[Ncol,K]^T  (+ bias)
// One wave computes a 16x64 strip (4 accumulators): the A fragment is loaded
// once per 32-wide K-step and reused against 4 B fragments -> operand traffic
// drops from 2KB/wmma to 1.25KB/wmma and 4 independent v_wmma issue per step.
// Requires Ncol % 64 == 0 (holds: 512/256/128) and M % 16 == 0 (10000 ok).
// Fragment packing follows CDNA5 ISA 16-bit A(16x32)/B(32x16) VGPR layouts.
// ---------------------------------------------------------------------------
__global__ void k_gemm_bf16_wmma(const unsigned short* __restrict__ A,
                                 const unsigned short* __restrict__ Bt,
                                 const float* __restrict__ bias,
                                 float* __restrict__ C,
                                 int mtiles, int K, int Ncol, int accumulate) {
  const int wave = (blockIdx.x * blockDim.x + threadIdx.x) >> 5;
  const int lane = threadIdx.x & 31;
  const int nstrips = Ncol >> 6;          // strips of 64 columns
  const int mt = wave / nstrips;
  const int ns = wave - mt * nstrips;
  if (mt >= mtiles) return;

  const int m0   = mt << 4;
  const int n0   = ns << 6;
  const int half = lane >> 4;     // 0: lanes 0-15, 1: lanes 16-31
  const int l15  = lane & 15;

  v8f acc[4];
#pragma unroll
  for (int t = 0; t < 4; ++t) {
    const int col = n0 + t * 16 + l15;
    const float binit = bias ? bias[col] : 0.0f;
    if (accumulate) {
#pragma unroll
      for (int r = 0; r < 8; ++r)
        acc[t][r] = C[(size_t)(m0 + r + half * 8) * Ncol + col] + binit;
    } else {
#pragma unroll
      for (int r = 0; r < 8; ++r) acc[t][r] = binit;
    }
  }

  const unsigned short* arow = A  + (size_t)(m0 + l15) * K;
  const unsigned short* brow = Bt + (size_t)(n0 + l15) * K;  // +t*16*K per tile

  for (int k0 = 0; k0 < K; k0 += 32) {
    __builtin_prefetch(arow + k0 + 32, 0, 3);
    union { v16bf v; unsigned int u[8]; } af;
#pragma unroll
    for (int v = 0; v < 8; ++v) {
      // A 16x32 bf16: VGPR v holds K pair (v>>2)*16 + half*8 + (v&3)*2
      const int ka = ((v >> 2) << 4) + (half << 3) + ((v & 3) << 1);
      af.u[v] = *(const unsigned int*)(arow + k0 + ka);
    }
#pragma unroll
    for (int t = 0; t < 4; ++t) {
      const unsigned short* bt = brow + (size_t)t * 16 * K + k0;
      __builtin_prefetch(bt + 32, 0, 3);
      union { v16bf v; unsigned int u[8]; } bf;
#pragma unroll
      for (int v = 0; v < 8; ++v) {
        // B 32x16 bf16: VGPR v holds K pair (v>>2)*16 + half*8 + (v&3)*2 (same as A)
        const int kb = ((v >> 2) << 4) + (half << 3) + ((v & 3) << 1);
        bf.u[v] = *(const unsigned int*)(bt + kb);
      }
      acc[t] = __builtin_amdgcn_wmma_f32_16x16x32_bf16(
          /*neg_a=*/false, af.v, /*neg_b=*/false, bf.v,
          /*c_mod=*/(short)0, acc[t], /*reuse_a=*/false, /*reuse_b=*/false);
    }
  }

#pragma unroll
  for (int t = 0; t < 4; ++t) {
    const int col = n0 + t * 16 + l15;
#pragma unroll
    for (int r = 0; r < 8; ++r)
      C[(size_t)(m0 + r + half * 8) * Ncol + col] = acc[t][r];
  }
}

// ---------------------------------------------------------------------------
// Attention score kernels
// ---------------------------------------------------------------------------
__global__ void k_attn_scores(const float* __restrict__ Hf,
                              const float* __restrict__ a_s,
                              const float* __restrict__ a_d,
                              float* __restrict__ es, float* __restrict__ ed,
                              int Nn, int Hh) {
  int i = blockIdx.x * blockDim.x + threadIdx.x;
  if (i >= Nn * Hh) return;
  int n = i / Hh, hh = i - n * Hh;
  const float* hp = Hf + (size_t)n * Hh * HID + (size_t)hh * HID;
  const float* as = a_s + hh * HID;
  const float* ad = a_d + hh * HID;
  float s1 = 0.f, s2 = 0.f;
  for (int c = 0; c < HID; ++c) {
    float v = hp[c];
    s1 += v * as[c];
    s2 += v * ad[c];
  }
  es[i] = s1;
  ed[i] = s2;
}

__global__ void k_init_max_denom(unsigned int* __restrict__ nmax,
                                 float* __restrict__ denom, int n) {
  int i = blockIdx.x * blockDim.x + threadIdx.x;
  if (i < n) { nmax[i] = 0x007FFFFFu /* sortable(-inf) */; denom[i] = 0.f; }
}

static __device__ __forceinline__ unsigned int f2sort(float f) {
  unsigned int u = __float_as_uint(f);
  return (u & 0x80000000u) ? ~u : (u | 0x80000000u);
}
static __device__ __forceinline__ float sort2f(unsigned int u) {
  unsigned int b = (u & 0x80000000u) ? (u & 0x7FFFFFFFu) : ~u;
  return __uint_as_float(b);
}

// leaky_relu(e_src[s]+e_dst[d]) and running segment-max over dst
__global__ void k_edge_alpha_max(const float* __restrict__ es,
                                 const float* __restrict__ ed,
                                 const int* __restrict__ srcA,
                                 const int* __restrict__ dstA,
                                 float* __restrict__ alpha,
                                 unsigned int* __restrict__ nmax,
                                 int E, int Etot, int Hh) {
  int e = blockIdx.x * blockDim.x + threadIdx.x;
  if (e >= Etot) return;
  int s = (e < E) ? srcA[e] : (e - E);
  int d = (e < E) ? dstA[e] : (e - E);
  for (int hh = 0; hh < Hh; ++hh) {
    float a = es[s * Hh + hh] + ed[d * Hh + hh];
    a = a > 0.f ? a : 0.2f * a;            // leaky_relu, slope 0.2
    alpha[(size_t)e * Hh + hh] = a;
    atomicMax(&nmax[d * Hh + hh], f2sort(a));
  }
}

// ex = exp(alpha - max[dst]); segment-sum into denom
__global__ void k_edge_exp_sum(float* __restrict__ alpha,
                               const unsigned int* __restrict__ nmax,
                               float* __restrict__ denom,
                               const int* __restrict__ dstA,
                               int E, int Etot, int Hh) {
  int e = blockIdx.x * blockDim.x + threadIdx.x;
  if (e >= Etot) return;
  int d = (e < E) ? dstA[e] : (e - E);
  for (int hh = 0; hh < Hh; ++hh) {
    float m  = sort2f(nmax[d * Hh + hh]);
    float ex = __expf(alpha[(size_t)e * Hh + hh] - m);
    alpha[(size_t)e * Hh + hh] = ex;
    atomicAdd(&denom[d * Hh + hh], ex);
  }
}

__global__ void k_init_out_bias(float* __restrict__ Of,
                                const float* __restrict__ bias, int Nn, int F) {
  int i = blockIdx.x * blockDim.x + threadIdx.x;
  if (i < Nn * F) Of[i] = bias[i % F];
}

// one wave per edge: Of[dst, h, :] += coeff[e,h] * Hf[src, h, :]
__global__ void k_edge_aggregate(const float* __restrict__ Hf,
                                 const float* __restrict__ ex,
                                 const float* __restrict__ denom,
                                 const int* __restrict__ srcA,
                                 const int* __restrict__ dstA,
                                 float* __restrict__ Of,
                                 int E, int Etot, int Hh) {
  int wave = (blockIdx.x * blockDim.x + threadIdx.x) >> 5;
  int lane = threadIdx.x & 31;
  if (wave >= Etot) return;
  int s = (wave < E) ? srcA[wave] : (wave - E);
  int d = (wave < E) ? dstA[wave] : (wave - E);
  const int F = Hh * HID;
  const float* hs = Hf + (size_t)s * F;
  float*       od = Of + (size_t)d * F;
  for (int j = lane; j < F; j += 32) {
    int hh = j >> 7;  // HID == 128
    float coeff = ex[(size_t)wave * Hh + hh] / (denom[d * Hh + hh] + 1e-16f);
    atomicAdd(&od[j], coeff * hs[j]);
  }
}

// ---------------------------------------------------------------------------
// Host orchestration
// ---------------------------------------------------------------------------
static inline int divup(int a, int b) { return (a + b - 1) / b; }

static void run_gat_layer(hipStream_t stream,
                          const unsigned short* actb, int K, int Hh,
                          const unsigned short* Wt, const float* bias,
                          const float* a_s, const float* a_d,
                          const int* srcA, const int* dstA,
                          float* Hf, float* Of, float* es, float* ed,
                          float* alpha, unsigned int* nmax, float* denom,
                          unsigned short* act_out) {
  const int F = Hh * HID;
  {
    int waves = MTILES * (F / 64);   // 16x64 strip per wave
    k_gemm_bf16_wmma<<<divup(waves * 32, 256), 256, 0, stream>>>(
        actb, Wt, nullptr, Hf, MTILES, K, F, 0);
  }
  {
    int n = N_NODES * Hh;
    k_attn_scores<<<divup(n, 256), 256, 0, stream>>>(Hf, a_s, a_d, es, ed,
                                                     N_NODES, Hh);
    k_init_max_denom<<<divup(n, 256), 256, 0, stream>>>(nmax, denom, n);
  }
  k_edge_alpha_max<<<divup(E_TOT, 256), 256, 0, stream>>>(
      es, ed, srcA, dstA, alpha, nmax, N_EDGES, E_TOT, Hh);
  k_edge_exp_sum<<<divup(E_TOT, 256), 256, 0, stream>>>(
      alpha, nmax, denom, dstA, N_EDGES, E_TOT, Hh);
  k_init_out_bias<<<divup(N_NODES * F, 256), 256, 0, stream>>>(Of, bias,
                                                               N_NODES, F);
  k_edge_aggregate<<<divup(E_TOT * 32, 256), 256, 0, stream>>>(
      Hf, alpha, denom, srcA, dstA, Of, N_EDGES, E_TOT, Hh);
  k_elu_cvt_bf16<<<divup(N_NODES * F, 256), 256, 0, stream>>>(Of, act_out,
                                                              N_NODES * F);
}

extern "C" void kernel_launch(void* const* d_in, const int* in_sizes, int n_in,
                              void* d_out, int out_size, void* d_ws,
                              size_t ws_size, hipStream_t stream) {
  const float* x   = (const float*)d_in[0];
  const float* W1  = (const float*)d_in[1];
  const float* a1s = (const float*)d_in[2];
  const float* a1d = (const float*)d_in[3];
  const float* b1  = (const float*)d_in[4];
  const float* W2  = (const float*)d_in[5];
  const float* a2s = (const float*)d_in[6];
  const float* a2d = (const float*)d_in[7];
  const float* b2  = (const float*)d_in[8];
  const float* W3  = (const float*)d_in[9];
  const float* a3s = (const float*)d_in[10];
  const float* a3d = (const float*)d_in[11];
  const float* b3  = (const float*)d_in[12];
  const float* Wl  = (const float*)d_in[13];
  const float* bl  = (const float*)d_in[14];
  const float* Wr  = (const float*)d_in[15];
  const float* br  = (const float*)d_in[16];
  const int*   ei  = (const int*)d_in[17];
  const int* srcA = ei;             // edge_index[0]
  const int* dstA = ei + N_EDGES;   // edge_index[1]
  float* out = (float*)d_out;

  // Workspace carve-up (256B aligned)
  char* ws = (char*)d_ws;
  size_t off = 0;
  auto carve = [&](size_t bytes) -> void* {
    void* p = ws + off;
    off += (bytes + 255) & ~(size_t)255;
    return p;
  };
  unsigned short* xb   = (unsigned short*)carve((size_t)N_NODES * 512 * 2);
  unsigned short* actb = (unsigned short*)carve((size_t)N_NODES * 512 * 2);
  float* Hf = (float*)carve((size_t)N_NODES * 512 * 4);
  float* Of = (float*)carve((size_t)N_NODES * 512 * 4);
  unsigned short* W1T = (unsigned short*)carve((size_t)512 * 512 * 2);
  unsigned short* W2T = (unsigned short*)carve((size_t)512 * 512 * 2);
  unsigned short* W3T = (unsigned short*)carve((size_t)128 * 512 * 2);
  unsigned short* WlT = (unsigned short*)carve((size_t)256 * 128 * 2);
  unsigned short* WrT = (unsigned short*)carve((size_t)256 * 512 * 2);
  float* es = (float*)carve((size_t)N_NODES * HEADS * 4);
  float* ed = (float*)carve((size_t)N_NODES * HEADS * 4);
  float* alpha = (float*)carve((size_t)E_TOT * HEADS * 4);
  unsigned int* nmax = (unsigned int*)carve((size_t)N_NODES * HEADS * 4);
  float* denom = (float*)carve((size_t)N_NODES * HEADS * 4);

  // One-shot weight conversions (fp32 -> transposed bf16)
  k_cvt_w_t<<<divup(512 * 512, 256), 256, 0, stream>>>(W1, W1T, 512, 512);
  k_cvt_w_t<<<divup(512 * 512, 256), 256, 0, stream>>>(W2, W2T, 512, 512);
  k_cvt_w_t<<<divup(512 * 128, 256), 256, 0, stream>>>(W3, W3T, 512, 128);
  k_cvt_w_t<<<divup(128 * 256, 256), 256, 0, stream>>>(Wl, WlT, 128, 256);
  k_cvt_w_t<<<divup(512 * 256, 256), 256, 0, stream>>>(Wr, WrT, 512, 256);
  k_cvt_bf16<<<divup(N_NODES * 512, 256), 256, 0, stream>>>(x, xb,
                                                            N_NODES * 512);

  // GAT layers (layer output, ELU'd + bf16'd, lands in actb)
  run_gat_layer(stream, xb,   512, HEADS, W1T, b1, a1s, a1d, srcA, dstA,
                Hf, Of, es, ed, alpha, nmax, denom, actb);
  run_gat_layer(stream, actb, 512, HEADS, W2T, b2, a2s, a2d, srcA, dstA,
                Hf, Of, es, ed, alpha, nmax, denom, actb);
  run_gat_layer(stream, actb, 512, 1,     W3T, b3, a3s, a3d, srcA, dstA,
                Hf, Of, es, ed, alpha, nmax, denom, actb);

  // Final: out = x@Wr + br   then   out += h3@Wl + bl
  {
    int waves = MTILES * (OUT_CH / 64);
    k_gemm_bf16_wmma<<<divup(waves * 32, 256), 256, 0, stream>>>(
        xb, WrT, br, out, MTILES, 512, OUT_CH, 0);
    k_gemm_bf16_wmma<<<divup(waves * 32, 256), 256, 0, stream>>>(
        actb, WlT, bl, out, MTILES, 128, OUT_CH, 1);
  }
  (void)in_sizes; (void)n_in; (void)out_size; (void)ws_size;
}
